// RecurrentBackbone_70652212019734
// MI455X (gfx1250) — matrix operand order repaired and checked by reading, not compiled, not run on hardware
//
#include <hip/hip_runtime.h>

// ---------- types for WMMA ----------
typedef __attribute__((ext_vector_type(16))) __bf16         v16bf;
typedef __attribute__((ext_vector_type(8)))  float          v8f;
typedef __attribute__((ext_vector_type(8)))  unsigned short ushort8;

union Frag {
  v16bf          v;
  ushort8        h[2];
  unsigned short s[16];
};

__device__ __forceinline__ unsigned short f32_to_bf16(float f) {
  unsigned int u = __float_as_uint(f);
  u += 0x7FFFu + ((u >> 16) & 1u);          // round-to-nearest-even
  return (unsigned short)(u >> 16);
}
__device__ __forceinline__ float bf16_to_f32(unsigned short h) {
  return __uint_as_float(((unsigned int)h) << 16);
}

// ---------- elementwise helpers ----------
__global__ void k_f32_to_bf16(const float* __restrict__ in,
                              unsigned short* __restrict__ out, int n) {
  int i = blockIdx.x * blockDim.x + threadIdx.x;
  if (i < n) out[i] = f32_to_bf16(in[i]);
}

__global__ void k_copy_f32(const float* __restrict__ in,
                           float* __restrict__ out, int n) {
  int i = blockIdx.x * blockDim.x + threadIdx.x;
  if (i < n) out[i] = in[i];
}

// ---------- generic bf16 GEMM:  C[M,N] = act(A[M,K] @ W[N,K]^T + bias) ----------
// block = 128 threads = 4 waves; wave tile = 16(M) x 64(N): 4 accumulators share
// one A fragment. Weights are L2-resident (192 MB L2); B loads go straight from
// global with no LDS staging -> no barriers / hard waits in the inner loop.
template <int K>
__global__ __launch_bounds__(128)
void k_gemm_bf16(const unsigned short* __restrict__ A,   // M x K bf16 row-major
                 const unsigned short* __restrict__ W,   // N x K bf16 row-major
                 const float* __restrict__ bias,         // N
                 float* __restrict__ outF,               // M x N f32 (or null)
                 unsigned short* __restrict__ outB,      // M x N bf16 (or null)
                 int N, int do_relu) {
  const int lane  = threadIdx.x & 31;
  const int wave  = threadIdx.x >> 5;
  const int n0    = blockIdx.y * 64;
  const int mBase = blockIdx.x * 64 + wave * 16;
  const int khalf = lane >> 4;                 // 0: lanes 0-15, 1: lanes 16-31
  const int nLane = lane & 15;
  const int mRow  = mBase + (lane & 15);

  // A fragment: lane holds row M=lane%16; lanes<16: K {0..7,16..23}, lanes>=16: {8..15,24..31}
  const unsigned short* ap0 = A + (size_t)mRow * K + khalf * 8;
  // B fragment: lane holds col N=lane%16; K {0..7,16..23} / {8..15,24..31} (same as A)
  const unsigned short* wp0 = W + (size_t)(n0 + nLane) * K + khalf * 8;

  v8f acc[4] = {v8f{}, v8f{}, v8f{}, v8f{}};

#pragma unroll 2
  for (int k0 = 0; k0 < K; k0 += 32) {
    __builtin_prefetch(ap0 + k0 + 128, 0, 1);       // global_prefetch_b8, A stream
    Frag a;
    a.h[0] = *(const ushort8*)(ap0 + k0);
    a.h[1] = *(const ushort8*)(ap0 + k0 + 16);
#pragma unroll
    for (int nt = 0; nt < 4; ++nt) {
      Frag b;
      const unsigned short* wp = wp0 + (size_t)nt * 16 * K + k0;
      b.h[0] = *(const ushort8*)(wp);
      b.h[1] = *(const ushort8*)(wp + 16);
      acc[nt] = __builtin_amdgcn_wmma_f32_16x16x32_bf16(false, a.v, false, b.v,
                                                        (short)0, acc[nt], false, false);
    }
  }

#pragma unroll
  for (int nt = 0; nt < 4; ++nt) {
    const int   nCol = n0 + nt * 16 + nLane;
    const float bv   = bias[nCol];
#pragma unroll
    for (int r = 0; r < 8; ++r) {
      int   m = mBase + r + 8 * khalf;             // C layout: VGPR r -> M=r / M=8+r
      float c = acc[nt][r] + bv;
      if (do_relu) c = fmaxf(c, 0.0f);
      size_t idx = (size_t)m * N + nCol;
      if (outF) outF[idx] = c;
      if (outB) outB[idx] = f32_to_bf16(c);
    }
  }
}

// ---------- LayerNorm over rows of width 512, one wave per row ----------
__global__ __launch_bounds__(128)
void k_layernorm(const float* __restrict__ in, const float* __restrict__ g,
                 const float* __restrict__ b, float* __restrict__ outF,
                 unsigned short* __restrict__ outB, int rows) {
  const int H   = 512;
  int lane = threadIdx.x & 31;
  int row  = blockIdx.x * 4 + (threadIdx.x >> 5);
  if (row >= rows) return;
  const float* rp = in + (size_t)row * H;

  float vals[16];
  float s = 0.f, sq = 0.f;
#pragma unroll
  for (int i = 0; i < 16; ++i) {
    float v = rp[lane + 32 * i];
    vals[i] = v; s += v; sq += v * v;
  }
#pragma unroll
  for (int m = 16; m >= 1; m >>= 1) {
    s  += __shfl_xor(s,  m, 32);
    sq += __shfl_xor(sq, m, 32);
  }
  float mean = s * (1.0f / H);
  float var  = sq * (1.0f / H) - mean * mean;
  float rstd = rsqrtf(var + 1e-5f);
#pragma unroll
  for (int i = 0; i < 16; ++i) {
    int   c = lane + 32 * i;
    float y = (vals[i] - mean) * rstd * g[c] + b[c];
    if (outF) outF[(size_t)row * H + c] = y;
    if (outB) outB[(size_t)row * H + c] = f32_to_bf16(y);
  }
}

// ---------- one GRU time step, fully fused ----------
// h kept in BOTH f32 (exact gate math / hT) and bf16 (WMMA A operand), ping-pong.
// Reset mask is exactly 0.0/1.0 and constant per row == per lane in the A-fragment
// layout, so masking the bf16 A fragment is a per-lane 8-dword select (no cvt).
__global__ __launch_bounds__(128)
void k_gru_step(const float* __restrict__ hprev32,        // N x H f32
                const unsigned short* __restrict__ hprev16, // N x H bf16
                const float* __restrict__ mt,             // N (on_reset row t, 0/1)
                const unsigned short* __restrict__ Whh,   // 3H x H bf16
                const float* __restrict__ bhh,            // 3H
                const unsigned short* __restrict__ gi_t,  // N x 3H bf16 (bih included)
                float* __restrict__ hnext32,              // N x H f32
                unsigned short* __restrict__ hnext16,     // N x H bf16
                float* __restrict__ ys,                   // T*N x H (d_out, pre-LN)
                int t) {
  const int H = 512, N = 512, G = 1536;
  const int lane  = threadIdx.x & 31;
  const int wave  = threadIdx.x >> 5;
  const int j0    = blockIdx.y * 64 + wave * 16;
  const int mBase = blockIdx.x * 16;
  const int khalf = lane >> 4;
  const int aRow  = mBase + (lane & 15);
  const int nOut  = j0 + (lane & 15);

  const bool keep = (mt[aRow] != 0.0f);        // whole lane's row shares one mask
  const unsigned short* hp0 = hprev16 + (size_t)aRow * H + khalf * 8;
  const unsigned short* wr0 = Whh + (size_t)nOut * H + khalf * 8;
  const ushort8 zero8 = {};

  v8f accR = {}, accZ = {}, accN = {};
#pragma unroll 2
  for (int k0 = 0; k0 < H; k0 += 32) {
    Frag a;
    ushort8 a0 = *(const ushort8*)(hp0 + k0);
    ushort8 a1 = *(const ushort8*)(hp0 + k0 + 16);
    a.h[0] = keep ? a0 : zero8;                // per-lane cndmask, no conversion
    a.h[1] = keep ? a1 : zero8;

    Frag br, bz, bn;
    const unsigned short* wr = wr0 + k0;
    const unsigned short* wz = wr + (size_t)H * H;   // +512 weight rows
    const unsigned short* wn = wz + (size_t)H * H;   // +1024 weight rows
    br.h[0] = *(const ushort8*)(wr); br.h[1] = *(const ushort8*)(wr + 16);
    bz.h[0] = *(const ushort8*)(wz); bz.h[1] = *(const ushort8*)(wz + 16);
    bn.h[0] = *(const ushort8*)(wn); bn.h[1] = *(const ushort8*)(wn + 16);

    accR = __builtin_amdgcn_wmma_f32_16x16x32_bf16(false, a.v, false, br.v,
                                                   (short)0, accR, false, false);
    accZ = __builtin_amdgcn_wmma_f32_16x16x32_bf16(false, a.v, false, bz.v,
                                                   (short)0, accZ, false, false);
    accN = __builtin_amdgcn_wmma_f32_16x16x32_bf16(false, a.v, false, bn.v,
                                                   (short)0, accN, false, false);
  }

  const float bR = bhh[nOut], bZ = bhh[nOut + 512], bN = bhh[nOut + 1024];
#pragma unroll
  for (int r = 0; r < 8; ++r) {
    int    nb     = mBase + r + 8 * khalf;
    size_t giBase = (size_t)nb * G + nOut;
    float ir  = bf16_to_f32(gi_t[giBase]);
    float iz  = bf16_to_f32(gi_t[giBase + 512]);
    float in_ = bf16_to_f32(gi_t[giBase + 1024]);
    float rr  = 1.0f / (1.0f + __expf(-(ir + accR[r] + bR)));
    float zz  = 1.0f / (1.0f + __expf(-(iz + accZ[r] + bZ)));
    float nn  = tanhf(in_ + rr * (accN[r] + bN));
    float hmv = hprev32[(size_t)nb * H + nOut] * mt[nb];   // exact f32 masked h
    float hnw = (1.0f - zz) * nn + zz * hmv;
    size_t hi = (size_t)nb * H + nOut;
    hnext32[hi] = hnw;
    hnext16[hi] = f32_to_bf16(hnw);            // convert ONCE per element here
    ys[((size_t)t * N + nb) * (size_t)H + nOut] = hnw;
  }
}

// ---------- host-side orchestration ----------
extern "C" void kernel_launch(void* const* d_in, const int* in_sizes, int n_in,
                              void* d_out, int out_size, void* d_ws, size_t ws_size,
                              hipStream_t stream) {
  (void)in_sizes; (void)n_in; (void)out_size; (void)ws_size;

  const float* obs      = (const float*)d_in[0];
  const float* hx       = (const float*)d_in[1];
  const float* on_reset = (const float*)d_in[2];
  const float* W1  = (const float*)d_in[3];
  const float* b1  = (const float*)d_in[4];
  const float* g1  = (const float*)d_in[5];
  const float* be1 = (const float*)d_in[6];
  const float* W2  = (const float*)d_in[7];
  const float* b2  = (const float*)d_in[8];
  const float* g2  = (const float*)d_in[9];
  const float* be2 = (const float*)d_in[10];
  const float* Wih = (const float*)d_in[11];
  const float* Whh = (const float*)d_in[12];
  const float* bih = (const float*)d_in[13];
  const float* bhh = (const float*)d_in[14];
  const float* gr  = (const float*)d_in[15];
  const float* br  = (const float*)d_in[16];

  const int T = 128, N = 512, H = 512, D = 128, G = 3 * H;
  const int M = T * N;  // 65536

  char*  ws  = (char*)d_ws;
  size_t off = 0;
  auto alloc = [&](size_t bytes) -> char* {
    char* p = ws + off;
    off += (bytes + 255) & ~(size_t)255;
    return p;
  };
  unsigned short* W1b   = (unsigned short*)alloc((size_t)H * D * 2);
  unsigned short* W2b   = (unsigned short*)alloc((size_t)H * H * 2);
  unsigned short* Wihb  = (unsigned short*)alloc((size_t)G * H * 2);
  unsigned short* Whhb  = (unsigned short*)alloc((size_t)G * H * 2);
  unsigned short* obsb  = (unsigned short*)alloc((size_t)M * D * 2);
  unsigned short* f1b   = (unsigned short*)alloc((size_t)M * H * 2);
  unsigned short* xb    = (unsigned short*)alloc((size_t)M * H * 2);
  float*          tmp   = (float*)         alloc((size_t)M * H * 4);
  unsigned short* gib   = (unsigned short*)alloc((size_t)M * G * 2);
  float*          hA32  = (float*)         alloc((size_t)N * H * 4);
  float*          hB32  = (float*)         alloc((size_t)N * H * 4);
  unsigned short* hA16  = (unsigned short*)alloc((size_t)N * H * 2);
  unsigned short* hB16  = (unsigned short*)alloc((size_t)N * H * 2);

  float* out = (float*)d_out;
  float* ysp = out;                      // (T*N, H) pre-LN then in-place LN
  float* hTp = out + (size_t)M * H;      // (N, H)

  const int thr = 256;
  // cast weights + obs to bf16
  k_f32_to_bf16<<<(H * D + thr - 1) / thr, thr, 0, stream>>>(W1,  W1b,  H * D);
  k_f32_to_bf16<<<(H * H + thr - 1) / thr, thr, 0, stream>>>(W2,  W2b,  H * H);
  k_f32_to_bf16<<<(G * H + thr - 1) / thr, thr, 0, stream>>>(Wih, Wihb, G * H);
  k_f32_to_bf16<<<(G * H + thr - 1) / thr, thr, 0, stream>>>(Whh, Whhb, G * H);
  k_f32_to_bf16<<<(M * D + thr - 1) / thr, thr, 0, stream>>>(obs, obsb, M * D);

  // layer 1: relu(obs @ W1^T + b1) -> LN -> bf16
  k_gemm_bf16<128><<<dim3(M / 64, H / 64), 128, 0, stream>>>(obsb, W1b, b1, tmp,
                                                             nullptr, H, 1);
  k_layernorm<<<M / 4, 128, 0, stream>>>(tmp, g1, be1, nullptr, f1b, M);

  // layer 2: relu(f1 @ W2^T + b2) -> LN -> bf16
  k_gemm_bf16<512><<<dim3(M / 64, H / 64), 128, 0, stream>>>(f1b, W2b, b2, tmp,
                                                             nullptr, H, 1);
  k_layernorm<<<M / 4, 128, 0, stream>>>(tmp, g2, be2, nullptr, xb, M);

  // gi = x @ Wih^T + bih  (bf16 workspace)
  k_gemm_bf16<512><<<dim3(M / 64, G / 64), 128, 0, stream>>>(xb, Wihb, bih, nullptr,
                                                             gib, G, 0);

  // h0 = hx (f32 + bf16 shadow)
  k_copy_f32   <<<(N * H + thr - 1) / thr, thr, 0, stream>>>(hx, hA32, N * H);
  k_f32_to_bf16<<<(N * H + thr - 1) / thr, thr, 0, stream>>>(hx, hA16, N * H);

  // sequential GRU scan, ping-pong h buffers
  for (int t = 0; t < T; ++t) {
    const float*          s32 = (t & 1) ? hB32 : hA32;
    const unsigned short* s16 = (t & 1) ? hB16 : hA16;
    float*                d32 = (t & 1) ? hA32 : hB32;
    unsigned short*       d16 = (t & 1) ? hA16 : hB16;
    k_gru_step<<<dim3(N / 16, H / 64), 128, 0, stream>>>(
        s32, s16, on_reset + (size_t)t * N, Whhb, bhh,
        gib + (size_t)t * N * G, d32, d16, ysp, t);
  }
  // T=128 even -> final h lands in hA32
  k_copy_f32<<<(N * H + thr - 1) / thr, thr, 0, stream>>>(hA32, hTp, N * H);
  k_layernorm<<<M / 4, 128, 0, stream>>>(ysp, gr, br, ysp, nullptr, M);
}
